// EvoformerBlock_53833120088080
// MI455X (gfx1250) — matrix operand order repaired and checked by reading, not compiled, not run on hardware
//
#include <hip/hip_runtime.h>
#include <stdint.h>

// ---------------------------------------------------------------------------
// CDNA5 (gfx1250) Evoformer block. wave32, WMMA 16x16x32 bf16, f32 accum.
// ---------------------------------------------------------------------------

typedef __attribute__((ext_vector_type(16))) __bf16 bf16x16;
typedef __attribute__((ext_vector_type(8)))  __bf16 bf16x8;
typedef __attribute__((ext_vector_type(8)))  float  f32x8;

#define ASYNC_K_STAGE 1   // use global_load_async_to_lds_b128 for K staging

#define DEVI __device__ __forceinline__

DEVI float rcpf(float x) { return __builtin_amdgcn_rcpf(x); }
DEVI float sigmoidf(float x) { return rcpf(1.0f + __expf(-x)); }

DEVI f32x8 wmma_bf16(bf16x16 a, bf16x16 b, f32x8 c) {
  return __builtin_amdgcn_wmma_f32_16x16x32_bf16(false, a, false, b, (short)0, c,
                                                 false, false);
}

// ---------------------------------------------------------------------------
// LayerNorm: one wave per row.
// ---------------------------------------------------------------------------
__global__ __launch_bounds__(256)
void ln_kernel(const float* __restrict__ x, const float* __restrict__ g,
               const float* __restrict__ b, float* __restrict__ out,
               int rows, int D) {
  int wave = (int)((blockIdx.x * blockDim.x + threadIdx.x) >> 5);
  int lane = threadIdx.x & 31;
  if (wave >= rows) return;
  const float* xr = x + (size_t)wave * D;
  float s = 0.f, ss = 0.f;
  for (int i = lane; i < D; i += 32) { float v = xr[i]; s += v; ss += v * v; }
  #pragma unroll
  for (int m = 16; m; m >>= 1) { s += __shfl_xor(s, m, 32); ss += __shfl_xor(ss, m, 32); }
  float invD = rcpf((float)D);            // D is a power of two -> exact
  float mu  = s * invD;
  float var = ss * invD - mu * mu;
  float rs  = rsqrtf(var + 1e-5f);
  float* orow = out + (size_t)wave * D;
  for (int i = lane; i < D; i += 32) orow[i] = (xr[i] - mu) * rs * g[i] + b[i];
}

// ---------------------------------------------------------------------------
// Transpose [A,B,C] -> [B,A,C]  and  dst[A,B,C] += src[B,A,C]
// ---------------------------------------------------------------------------
__global__ void transpose_kernel(const float* __restrict__ src, float* __restrict__ dst,
                                 int A, int B, int C) {
  size_t idx = (size_t)blockIdx.x * blockDim.x + threadIdx.x;
  size_t total = (size_t)A * B * C;
  if (idx >= total) return;
  int c = (int)(idx % C);
  size_t ij = idx / C;
  int j = (int)(ij % B);
  int i = (int)(ij / B);
  dst[((size_t)j * A + i) * C + c] = src[idx];
}

__global__ void addtrans_kernel(float* __restrict__ dst, const float* __restrict__ src,
                                int A, int B, int C) {
  size_t idx = (size_t)blockIdx.x * blockDim.x + threadIdx.x;
  size_t total = (size_t)A * B * C;
  if (idx >= total) return;
  int c = (int)(idx % C);
  size_t ij = idx / C;
  int j = (int)(ij % B);
  int i = (int)(ij / B);
  dst[idx] += src[((size_t)j * A + i) * C + c];
}

// ---------------------------------------------------------------------------
// Mask bias: out[n*L+k] = (mask - 1) * 1e9  (optionally transposed read)
// ---------------------------------------------------------------------------
__global__ void maskbias_kernel(const float* __restrict__ mask, float* __restrict__ out,
                                int N, int L, int transposed) {
  int idx = (int)(blockIdx.x * blockDim.x + threadIdx.x);
  if (idx >= N * L) return;
  int n = idx / L, k = idx % L;
  float mv = transposed ? mask[(size_t)k * N + n] : mask[(size_t)n * L + k];
  out[idx] = (mv - 1.0f) * 1e9f;
}

// ---------------------------------------------------------------------------
// Bias projection: out[h][i][j] = sum_d t[(i*R+j)*D+d] * wb[d*H+h]   (H = 8 or 4)
// ---------------------------------------------------------------------------
__global__ void biasproj_kernel(const float* __restrict__ t, const float* __restrict__ wb,
                                float* __restrict__ out, int R, int D, int H) {
  int idx = (int)(blockIdx.x * blockDim.x + threadIdx.x);
  int total = H * R * R;
  if (idx >= total) return;
  int h = idx / (R * R);
  int ij = idx % (R * R);
  const float* tr = t + (size_t)ij * D;
  float s = 0.f;
  for (int d = 0; d < D; d += 4) {
    float4 tv = *(const float4*)(tr + d);
    s += tv.x * wb[(d + 0) * H + h];
    s += tv.y * wb[(d + 1) * H + h];
    s += tv.z * wb[(d + 2) * H + h];
    s += tv.w * wb[(d + 3) * H + h];
  }
  out[idx] = s;
}

// ---------------------------------------------------------------------------
// Generic GEMM:  out[rows,P] = epilogue(A[rows,D] @ W[D,P])
// Block tile 64x32, 8 waves (4x2 grid of 16x16 WMMA tiles), K-step 32.
// epilogue: v = act(acc + bias[col]); v *= mul[r,c]; v *= rowscale[r]; v += res[r,c]
// obf != 0 -> write output as bf16 (reinterpreting out as __bf16*)
// ---------------------------------------------------------------------------
__global__ __launch_bounds__(256)
void gemm_kernel(const float* __restrict__ A, const float* __restrict__ W,
                 const float* __restrict__ bias, float* __restrict__ out,
                 int rows, int D, int P, int act,
                 const float* __restrict__ rowscale,
                 const float* __restrict__ mulbuf,
                 const float* __restrict__ residual, int obf) {
  __shared__ __bf16 lA[64 * 32];
  __shared__ __bf16 lB[32 * 32];
  int tid  = threadIdx.x;
  int lane = tid & 31, wv = tid >> 5;
  int mi = wv & 3, ni = wv >> 2;
  int half = lane >> 4, l16 = lane & 15;
  int row0 = blockIdx.x * 64;
  int n0   = blockIdx.y * 32;
  // per-thread staging coordinates
  int ar  = tid >> 2;               // A row 0..63
  int akk = (tid & 3) * 8;          // A k-offset {0,8,16,24}
  int wkk = tid >> 3;               // W k-row 0..31
  int wn4 = (tid & 7) * 4;          // W n-offset {0..28}
  f32x8 acc = {};
  for (int k0 = 0; k0 < D; k0 += 32) {
    __syncthreads();
    {
      const float* ap = A + (size_t)(row0 + ar) * D + k0 + akk;
      float4 a0 = *(const float4*)(ap);
      float4 a1 = *(const float4*)(ap + 4);
      int o = ar * 32 + akk;
      lA[o + 0] = (__bf16)a0.x; lA[o + 1] = (__bf16)a0.y;
      lA[o + 2] = (__bf16)a0.z; lA[o + 3] = (__bf16)a0.w;
      lA[o + 4] = (__bf16)a1.x; lA[o + 5] = (__bf16)a1.y;
      lA[o + 6] = (__bf16)a1.z; lA[o + 7] = (__bf16)a1.w;
      if (k0 + 32 < D) __builtin_prefetch(ap + 32, 0, 3);   // global_prefetch_b8
      const float* wp = W + (size_t)(k0 + wkk) * P + n0 + wn4;
      float4 w0 = *(const float4*)(wp);
      lB[(wn4 + 0) * 32 + wkk] = (__bf16)w0.x;
      lB[(wn4 + 1) * 32 + wkk] = (__bf16)w0.y;
      lB[(wn4 + 2) * 32 + wkk] = (__bf16)w0.z;
      lB[(wn4 + 3) * 32 + wkk] = (__bf16)w0.w;
    }
    __syncthreads();
    bf16x16 af, bfr;
    int am = mi * 16 + l16;
    #pragma unroll
    for (int e = 0; e < 16; ++e) {
      int kk = (half ? 8 : 0) + (e >> 3) * 16 + (e & 7);
      af[e] = lA[am * 32 + kk];
    }
    int bn = ni * 16 + l16;
    #pragma unroll
    for (int e = 0; e < 16; ++e) {
      int kk = (half ? 8 : 0) + (e >> 3) * 16 + (e & 7);
      bfr[e] = lB[bn * 32 + kk];
    }
    acc = wmma_bf16(af, bfr, acc);
  }
  int col = n0 + ni * 16 + l16;
  float bv = bias ? bias[col] : 0.f;
  #pragma unroll
  for (int gI = 0; gI < 8; ++gI) {
    int row = row0 + mi * 16 + gI + (half ? 8 : 0);
    float v = acc[gI] + bv;
    if (act == 1)       v = v > 0.f ? v : 0.f;
    else if (act == 2)  v = sigmoidf(v);
    if (mulbuf)   v *= mulbuf[(size_t)row * P + col];
    if (rowscale) v *= rowscale[row];
    if (residual) v += residual[(size_t)row * P + col];
    if (obf) ((__bf16*)out)[(size_t)row * P + col] = (__bf16)v;
    else     out[(size_t)row * P + col] = v;
  }
}

// ---------------------------------------------------------------------------
// Fused gated attention. One block per (n, h). head dim c=32.
// Q,K,V : bf16 [N, L, H*32];  G : f32 [N, L, H*32];  maskb : [N,L];
// pairb : [H,L,L] or null.
// out   : f32 [N, L, H*32] = sigmoid(G) * softmax(QK^T*scale + biases) V
// Dynamic LDS layout (base offset 0): ldsK [L*32] | ldsVT [32*L] | ldsP [8*512]
// ---------------------------------------------------------------------------
template <int L>
__global__ __launch_bounds__(256)
void attn_kernel(const __bf16* __restrict__ Q, const __bf16* __restrict__ K,
                 const __bf16* __restrict__ V, const float* __restrict__ G,
                 const float* __restrict__ maskb, const float* __restrict__ pairb,
                 float* __restrict__ out, int H, float scale) {
  constexpr int L16 = L / 16;
  constexpr int L32 = L / 32;
  constexpr int QTPW = (L16 + 7) / 8;   // q-tiles per wave (8 waves)
  extern __shared__ __bf16 smem[];
  __bf16* ldsK  = smem;                  // [key][c], byte offset 0
  __bf16* ldsVT = smem + L * 32;         // [c][key]
  __bf16* ldsP  = smem + 2 * L * 32;     // per-wave 16x32 chunks
  int n = blockIdx.x, h = blockIdx.y;
  int HC = H * 32;
  int tid = threadIdx.x, lane = tid & 31, wv = tid >> 5;
  int half = lane >> 4, l16 = lane & 15;
  const size_t base = ((size_t)n * L) * HC + h * 32;

  // stage K (async DMA straight to LDS; no conversion needed: bf16 in, bf16 out)
  for (int idx = tid * 8; idx < L * 32; idx += 2048) {
    int l = idx >> 5, c = idx & 31;
    const __bf16* kp = K + base + (size_t)l * HC + c;
#if ASYNC_K_STAGE
    unsigned ldsoff = (unsigned)(idx * 2);          // byte offset of ldsK[idx]
    asm volatile("global_load_async_to_lds_b128 %0, %1, off"
                 :: "v"(ldsoff), "v"(kp) : "memory");
#else
    *(bf16x8*)(ldsK + idx) = *(const bf16x8*)kp;
#endif
    bf16x8 vv = *(const bf16x8*)(V + base + (size_t)l * HC + c);
    #pragma unroll
    for (int u = 0; u < 8; ++u) ldsVT[(c + u) * L + l] = vv[u];
  }
#if ASYNC_K_STAGE
  asm volatile("s_wait_asynccnt 0" ::: "memory");
#endif
  __syncthreads();
  __bf16* ldsPw = ldsP + wv * (16 * 32);

  for (int qi = 0; qi < QTPW; ++qi) {
    int qt = wv + 8 * qi;
    // Q fragment (A layout) -- raw bf16, scale applied to scores later
    bf16x16 aq;
    int qm = qt * 16 + l16;
    {
      const __bf16* qp = Q + base + (size_t)qm * HC + (half ? 8 : 0);
      bf16x8 qlo = *(const bf16x8*)(qp);
      bf16x8 qhi = *(const bf16x8*)(qp + 16);
      #pragma unroll
      for (int u = 0; u < 8; ++u) { aq[u] = qlo[u]; aq[8 + u] = qhi[u]; }
    }
    // scores: 16 rows x L keys, kept in registers as L16 C-tiles
    f32x8 sc[L16];
    #pragma unroll
    for (int kb = 0; kb < L16; ++kb) {
      bf16x16 bk;
      #pragma unroll
      for (int e = 0; e < 16; ++e) {
        int kk = (half ? 8 : 0) + (e >> 3) * 16 + (e & 7);
        bk[e] = ldsK[(kb * 16 + l16) * 32 + kk];
      }
      f32x8 zc = {};
      sc[kb] = wmma_bf16(aq, bk, zc);
      int key = kb * 16 + l16;
      float mb = maskb[(size_t)n * L + key];
      #pragma unroll
      for (int gI = 0; gI < 8; ++gI) {
        float pb = 0.f;
        if (pairb) {
          int q = qt * 16 + gI + (half ? 8 : 0);
          pb = pairb[((size_t)h * L + q) * L + key];
        }
        sc[kb][gI] = sc[kb][gI] * scale + mb + pb;
      }
    }
    // softmax per row (rows live across the 16-lane half)
    float rinv[8];
    #pragma unroll
    for (int gI = 0; gI < 8; ++gI) {
      float mx = -1e30f;
      #pragma unroll
      for (int kb = 0; kb < L16; ++kb) mx = fmaxf(mx, sc[kb][gI]);
      #pragma unroll
      for (int m = 1; m < 16; m <<= 1) mx = fmaxf(mx, __shfl_xor(mx, m, 32));
      float sm = 0.f;
      #pragma unroll
      for (int kb = 0; kb < L16; ++kb) {
        float e = __expf(sc[kb][gI] - mx);
        sc[kb][gI] = e; sm += e;
      }
      #pragma unroll
      for (int m = 1; m < 16; m <<= 1) sm += __shfl_xor(sm, m, 32);
      rinv[gI] = rcpf(sm);
    }
    // P @ V via 32-key chunks (P transposed through wave-local LDS chunk)
    f32x8 o0 = {}, o1 = {};
    #pragma unroll
    for (int kb2 = 0; kb2 < L32; ++kb2) {
      #pragma unroll
      for (int bI = 0; bI < 2; ++bI) {
        int kb = kb2 * 2 + bI;
        #pragma unroll
        for (int gI = 0; gI < 8; ++gI) {
          int rl = gI + (half ? 8 : 0);
          ldsPw[rl * 32 + bI * 16 + l16] = (__bf16)sc[kb][gI];
        }
      }
      bf16x16 ap;
      #pragma unroll
      for (int e = 0; e < 16; ++e) {
        int kk = (half ? 8 : 0) + (e >> 3) * 16 + (e & 7);
        ap[e] = ldsPw[l16 * 32 + kk];
      }
      bf16x16 bv0, bv1;
      #pragma unroll
      for (int e = 0; e < 16; ++e) {
        int key = kb2 * 32 + (half ? 8 : 0) + (e >> 3) * 16 + (e & 7);
        bv0[e] = ldsVT[l16 * L + key];
        bv1[e] = ldsVT[(16 + l16) * L + key];
      }
      o0 = wmma_bf16(ap, bv0, o0);
      o1 = wmma_bf16(ap, bv1, o1);
    }
    // normalize, gate, store
    #pragma unroll
    for (int gI = 0; gI < 8; ++gI) {
      int q = qt * 16 + gI + (half ? 8 : 0);
      size_t i0 = base + (size_t)q * HC + l16;
      float g0 = sigmoidf(G[i0]);
      float g1 = sigmoidf(G[i0 + 16]);
      out[i0]      = o0[gI] * rinv[gI] * g0;
      out[i0 + 16] = o1[gI] * rinv[gI] * g1;
    }
  }
}

// ---------------------------------------------------------------------------
// Fused outer-product-mean. One block per (i, 16 j's).
// Phase 1: 16 outer blocks (32c x 32d over S) via WMMA -> LDS (bf16).
// Phase 2: [16 pairs x 1024] @ wo[1024,128] via WMMA, scaled by 1/(eps+norm),
//          residual into z.
// a,b: [S, R, 32];  mask: [S,R];  z: [R,R,128]
// ---------------------------------------------------------------------------
__global__ __launch_bounds__(256)
void opm_kernel(const float* __restrict__ a, const float* __restrict__ b,
                const float* __restrict__ mask, const float* __restrict__ wo,
                const float* __restrict__ bo, float* __restrict__ zio,
                int R, int S) {
  __shared__ __bf16 lAs[128 * 32];     // a[:,i,:] staged (S=128)
  __shared__ __bf16 lOut[16 * 1024];   // [pair][c*32+d]
  __shared__ float lInv[16];
  int i = blockIdx.x, j0 = blockIdx.y * 16;
  int tid = threadIdx.x, lane = tid & 31, wv = tid >> 5;
  int half = lane >> 4, l16 = lane & 15;
  // stage a[:,i,:] (S x 32) into LDS
  #pragma unroll
  for (int t = 0; t < 4; ++t) {
    int idx = (tid + t * 256) * 4;     // 0..4095 step 4
    int s = idx >> 5, c = idx & 31;
    float4 av = *(const float4*)(a + ((size_t)s * R + i) * 32 + c);
    lAs[s * 32 + c + 0] = (__bf16)av.x;
    lAs[s * 32 + c + 1] = (__bf16)av.y;
    lAs[s * 32 + c + 2] = (__bf16)av.z;
    lAs[s * 32 + c + 3] = (__bf16)av.w;
  }
  // norms: 16 threads per pair (lanes of a half-wave share one pair)
  {
    int p = tid >> 4;          // 0..15
    int s0 = tid & 15;
    float ns = 0.f;
    for (int s = s0; s < S; s += 16)
      ns += mask[(size_t)s * R + i] * mask[(size_t)s * R + (j0 + p)];
    #pragma unroll
    for (int m = 1; m < 16; m <<= 1) ns += __shfl_xor(ns, m, 32);
    if (s0 == 0) lInv[p] = rcpf(1e-3f + ns);
  }
  __syncthreads();
  // phase 1: each wave computes two pairs' 32x32 outer blocks
  #pragma unroll
  for (int pi = 0; pi < 2; ++pi) {
    int p = wv * 2 + pi;
    int j = j0 + p;
    f32x8 acc[2][2] = {};
    for (int k0 = 0; k0 < S; k0 += 32) {
      bf16x16 af[2], bfr[2];
      #pragma unroll
      for (int mi = 0; mi < 2; ++mi)
        #pragma unroll
        for (int e = 0; e < 16; ++e) {
          int s = k0 + (half ? 8 : 0) + (e >> 3) * 16 + (e & 7);
          af[mi][e] = lAs[s * 32 + mi * 16 + l16];
        }
      #pragma unroll
      for (int ni = 0; ni < 2; ++ni)
        #pragma unroll
        for (int e = 0; e < 16; ++e) {
          int s = k0 + (half ? 8 : 0) + (e >> 3) * 16 + (e & 7);
          bfr[ni][e] = (__bf16)b[((size_t)s * R + j) * 32 + ni * 16 + l16];
        }
      #pragma unroll
      for (int mi = 0; mi < 2; ++mi)
        #pragma unroll
        for (int ni = 0; ni < 2; ++ni)
          acc[mi][ni] = wmma_bf16(af[mi], bfr[ni], acc[mi][ni]);
    }
    #pragma unroll
    for (int mi = 0; mi < 2; ++mi)
      #pragma unroll
      for (int ni = 0; ni < 2; ++ni)
        #pragma unroll
        for (int gI = 0; gI < 8; ++gI) {
          int c = mi * 16 + gI + (half ? 8 : 0);
          int d = ni * 16 + l16;
          lOut[p * 1024 + c * 32 + d] = (__bf16)acc[mi][ni][gI];
        }
  }
  __syncthreads();
  // phase 2: [16 x 1024] @ wo -> wave wv owns output cols [wv*16, wv*16+16)
  f32x8 acc2 = {};
  for (int k0 = 0; k0 < 1024; k0 += 32) {
    bf16x16 af, bfr;
    #pragma unroll
    for (int e = 0; e < 16; ++e) {
      int kk = k0 + (half ? 8 : 0) + (e >> 3) * 16 + (e & 7);
      af[e] = lOut[l16 * 1024 + kk];
    }
    #pragma unroll
    for (int e = 0; e < 16; ++e) {
      int kk = k0 + (half ? 8 : 0) + (e >> 3) * 16 + (e & 7);
      bfr[e] = (__bf16)wo[(size_t)kk * 128 + wv * 16 + l16];
    }
    acc2 = wmma_bf16(af, bfr, acc2);
  }
  int col = wv * 16 + l16;
  float bv = bo[col];
  #pragma unroll
  for (int gI = 0; gI < 8; ++gI) {
    int p = gI + (half ? 8 : 0);
    size_t zr = ((size_t)i * R + (j0 + p)) * 128 + col;
    zio[zr] += lInv[p] * (acc2[gI] + bv);
  }
}

// ---------------------------------------------------------------------------
// Triangle-multiply einsum, batched over channel c:
//   outgoing: x[i,j,c] = sum_k a[i,k,c]*b[j,k,c]
//   incoming: x[i,j,c] = sum_k a[k,i,c]*b[k,j,c]
// ---------------------------------------------------------------------------
__global__ __launch_bounds__(128)
void trimul_kernel(const float* __restrict__ a, const float* __restrict__ bsrc,
                   float* __restrict__ x, int R, int C, int outgoing) {
  __shared__ __bf16 lA[32 * 32];
  __shared__ __bf16 lB[32 * 32];
  int c = blockIdx.z;
  int i0 = blockIdx.x * 32, j0 = blockIdx.y * 32;
  int tid = threadIdx.x, lane = tid & 31, wv = tid >> 5;
  int half = lane >> 4, l16 = lane & 15;
  int mi = wv & 1, ni = wv >> 1;
  f32x8 acc = {};
  for (int k0 = 0; k0 < R; k0 += 32) {
    __syncthreads();
    #pragma unroll
    for (int t = 0; t < 8; ++t) {
      int idx = tid + t * 128;          // 0..1023
      int r = idx >> 5, kk = idx & 31;
      size_t ia = outgoing ? ((size_t)(i0 + r) * R + (k0 + kk))
                           : ((size_t)(k0 + kk) * R + (i0 + r));
      size_t ib = outgoing ? ((size_t)(j0 + r) * R + (k0 + kk))
                           : ((size_t)(k0 + kk) * R + (j0 + r));
      lA[idx] = (__bf16)a[ia * C + c];
      lB[idx] = (__bf16)bsrc[ib * C + c];
    }
    __syncthreads();
    bf16x16 af, bfr;
    int am = mi * 16 + l16;
    #pragma unroll
    for (int e = 0; e < 16; ++e) {
      int kk = (half ? 8 : 0) + (e >> 3) * 16 + (e & 7);
      af[e] = lA[am * 32 + kk];
    }
    int bn = ni * 16 + l16;
    #pragma unroll
    for (int e = 0; e < 16; ++e) {
      int kk = (half ? 8 : 0) + (e >> 3) * 16 + (e & 7);
      bfr[e] = lB[bn * 32 + kk];
    }
    acc = wmma_bf16(af, bfr, acc);
  }
  #pragma unroll
  for (int gI = 0; gI < 8; ++gI) {
    int ig = i0 + mi * 16 + gI + (half ? 8 : 0);
    int jg = j0 + ni * 16 + l16;
    x[((size_t)ig * R + jg) * C + c] = acc[gI];
  }
}

// ---------------------------------------------------------------------------
// Host orchestration
// ---------------------------------------------------------------------------
namespace {

// flattened param indices (JAX pytree: sorted dict keys, DFS)
enum {
  I_M = 0, I_Z, I_MSA_MASK, I_PAIR_MASK,
  C_BG, C_BO, C_LNB, C_LNG, C_WG, C_WK, C_WO, C_WQ, C_WV,                  // col
  MT_B1, MT_B2, MT_LNB, MT_LNG, MT_W1, MT_W2,                              // msa_t
  O_BA, O_BB, O_BO, O_LNB, O_LNG, O_WA, O_WB, O_WO,                        // opm
  PT_B1, PT_B2, PT_LNB, PT_LNG, PT_W1, PT_W2,                              // pair_t
  R_BG, R_BO, R_LNMB, R_LNMG, R_LNZB, R_LNZG, R_WB, R_WG, R_WK, R_WO, R_WQ, R_WV, // row
  TAE_BG, TAE_BO, TAE_LNB, TAE_LNG, TAE_WB, TAE_WG, TAE_WK, TAE_WO, TAE_WQ, TAE_WV,
  TAS_BG, TAS_BO, TAS_LNB, TAS_LNG, TAS_WB, TAS_WG, TAS_WK, TAS_WO, TAS_WQ, TAS_WV,
  TMI_BAG, TMI_BAP, TMI_BBG, TMI_BBP, TMI_BG, TMI_BO, TMI_LNINB, TMI_LNING,
  TMI_LNOUTB, TMI_LNOUTG, TMI_WAG, TMI_WAP, TMI_WBG, TMI_WBP, TMI_WG, TMI_WO,
  TMO_BAG, TMO_BAP, TMO_BBG, TMO_BBP, TMO_BG, TMO_BO, TMO_LNINB, TMO_LNING,
  TMO_LNOUTB, TMO_LNOUTG, TMO_WAG, TMO_WAP, TMO_WBG, TMO_WBP, TMO_WG, TMO_WO,
};

constexpr int S_ = 128, R_ = 256, CM_ = 256, CZ_ = 128;
constexpr size_t ME = (size_t)8388608;   // 1*128*256*256 == 1*256*256*128
constexpr float SCL = 0.17677669529663687f;  // 1/sqrt(32)

void launch_ln(hipStream_t st, const float* x, const float* g, const float* b,
               float* out, int rows, int D) {
  ln_kernel<<<dim3(rows / 8), dim3(256), 0, st>>>(x, g, b, out, rows, D);
}
void launch_gemm(hipStream_t st, const float* A, const float* W, const float* bias,
                 float* out, int rows, int D, int P, int act,
                 const float* rowscale, const float* mulbuf, const float* residual,
                 int obf = 0) {
  gemm_kernel<<<dim3(rows / 64, P / 32), dim3(256), 0, st>>>(
      A, W, bias, out, rows, D, P, act, rowscale, mulbuf, residual, obf);
}

constexpr size_t attn_lds(int L) { return (size_t)(2 * L * 32 + 8 * 16 * 32) * 2; }

}  // namespace

extern "C" void kernel_launch(void* const* d_in, const int* in_sizes, int n_in,
                              void* d_out, int out_size, void* d_ws, size_t ws_size,
                              hipStream_t stream) {
  (void)in_sizes; (void)n_in; (void)out_size; (void)ws_size;
  auto in = [&](int i) { return (const float*)d_in[i]; };
  float* W = (float*)d_ws;
  auto buf = [&](int i) { return W + (size_t)i * ME; };

  float* m_cur = buf(0);
  float* z_cur = buf(1);
  float* t0    = buf(2);
  float* t1    = buf(3);
  float* q     = buf(4);
  float* k     = buf(5);
  float* v     = buf(6);
  float* g     = buf(7);
  float* tA    = buf(8);
  float* tB    = buf(9);
  float* hidden = buf(4);               // overlaps q..g (safe: disjoint phases)
  float* pairb = buf(10);               // up to 8*256*256 = 524288
  float* maskb = buf(10) + 524288;      // up to 65536

  const __bf16* qb = (const __bf16*)q;
  const __bf16* kb = (const __bf16*)k;
  const __bf16* vb = (const __bf16*)v;

  const float* msa_mask  = in(I_MSA_MASK);   // [128,256]
  const float* pair_mask = in(I_PAIR_MASK);  // [256,256]

  hipMemcpyAsync(m_cur, d_in[I_M], ME * sizeof(float), hipMemcpyDeviceToDevice, stream);
  hipMemcpyAsync(z_cur, d_in[I_Z], ME * sizeof(float), hipMemcpyDeviceToDevice, stream);

  // ---------------- Row attention (pair bias) ----------------
  launch_ln(stream, m_cur, in(R_LNMG), in(R_LNMB), t0, S_ * R_, CM_);
  launch_ln(stream, z_cur, in(R_LNZG), in(R_LNZB), t1, R_ * R_, CZ_);
  biasproj_kernel<<<dim3((8 * R_ * R_) / 256), dim3(256), 0, stream>>>(
      t1, in(R_WB), pairb, R_, CZ_, 8);
  maskbias_kernel<<<dim3((S_ * R_) / 256), dim3(256), 0, stream>>>(
      msa_mask, maskb, S_, R_, 0);
  launch_gemm(stream, t0, in(R_WQ), nullptr, q, S_ * R_, CM_, 256, 0, nullptr, nullptr, nullptr, 1);
  launch_gemm(stream, t0, in(R_WK), nullptr, k, S_ * R_, CM_, 256, 0, nullptr, nullptr, nullptr, 1);
  launch_gemm(stream, t0, in(R_WV), nullptr, v, S_ * R_, CM_, 256, 0, nullptr, nullptr, nullptr, 1);
  launch_gemm(stream, t0, in(R_WG), in(R_BG), g, S_ * R_, CM_, 256, 0, nullptr, nullptr, nullptr);
  attn_kernel<256><<<dim3(S_, 8), dim3(256), attn_lds(256), stream>>>(
      qb, kb, vb, g, maskb, pairb, tA, 8, SCL);
  launch_gemm(stream, tA, in(R_WO), in(R_BO), m_cur, S_ * R_, 256, CM_, 0, nullptr, nullptr, m_cur);

  // ---------------- Column attention ----------------
  launch_ln(stream, m_cur, in(C_LNG), in(C_LNB), t0, S_ * R_, CM_);
  transpose_kernel<<<dim3((unsigned)(ME / 256)), dim3(256), 0, stream>>>(t0, t1, S_, R_, CM_);
  maskbias_kernel<<<dim3((R_ * S_) / 256), dim3(256), 0, stream>>>(msa_mask, maskb, R_, S_, 1);
  launch_gemm(stream, t1, in(C_WQ), nullptr, q, R_ * S_, CM_, 256, 0, nullptr, nullptr, nullptr, 1);
  launch_gemm(stream, t1, in(C_WK), nullptr, k, R_ * S_, CM_, 256, 0, nullptr, nullptr, nullptr, 1);
  launch_gemm(stream, t1, in(C_WV), nullptr, v, R_ * S_, CM_, 256, 0, nullptr, nullptr, nullptr, 1);
  launch_gemm(stream, t1, in(C_WG), in(C_BG), g, R_ * S_, CM_, 256, 0, nullptr, nullptr, nullptr);
  attn_kernel<128><<<dim3(R_, 8), dim3(256), attn_lds(128), stream>>>(
      qb, kb, vb, g, maskb, nullptr, tA, 8, SCL);
  launch_gemm(stream, tA, in(C_WO), in(C_BO), tB, R_ * S_, 256, CM_, 0, nullptr, nullptr, nullptr);
  addtrans_kernel<<<dim3((unsigned)(ME / 256)), dim3(256), 0, stream>>>(m_cur, tB, S_, R_, CM_);

  // ---------------- MSA transition ----------------
  launch_ln(stream, m_cur, in(MT_LNG), in(MT_LNB), t0, S_ * R_, CM_);
  launch_gemm(stream, t0, in(MT_W1), in(MT_B1), hidden, S_ * R_, CM_, 1024, 1, nullptr, nullptr, nullptr);
  launch_gemm(stream, hidden, in(MT_W2), in(MT_B2), m_cur, S_ * R_, 1024, CM_, 0, nullptr, nullptr, m_cur);

  // ---------------- Outer product mean ----------------
  launch_ln(stream, m_cur, in(O_LNG), in(O_LNB), t0, S_ * R_, CM_);
  launch_gemm(stream, t0, in(O_WA), in(O_BA), tA, S_ * R_, CM_, 32, 0, msa_mask, nullptr, nullptr);
  launch_gemm(stream, t0, in(O_WB), in(O_BB), tB, S_ * R_, CM_, 32, 0, msa_mask, nullptr, nullptr);
  opm_kernel<<<dim3(R_, R_ / 16), dim3(256), 0, stream>>>(
      tA, tB, msa_mask, in(O_WO), in(O_BO), z_cur, R_, S_);

  // ---------------- Triangle multiply (outgoing, then incoming) ----------------
  for (int dir = 0; dir < 2; ++dir) {
    const int P0 = dir ? TMI_BAG : TMO_BAG;   // base of the 16-entry group
    auto pp = [&](int rel) { return in(P0 + rel); };
    // rel: 0 bag,1 bap,2 bbg,3 bbp,4 bg,5 bo,6 ln_in_b,7 ln_in_g,8 ln_out_b,
    //      9 ln_out_g,10 wag,11 wap,12 wbg,13 wbp,14 wg,15 wo
    launch_ln(stream, z_cur, pp(7), pp(6), t0, R_ * R_, CZ_);
    launch_gemm(stream, t0, pp(11), pp(1), q, R_ * R_, CZ_, 128, 0, nullptr, nullptr, nullptr);
    launch_gemm(stream, t0, pp(10), pp(0), tA, R_ * R_, CZ_, 128, 2, pair_mask, q, nullptr);
    launch_gemm(stream, t0, pp(13), pp(3), k, R_ * R_, CZ_, 128, 0, nullptr, nullptr, nullptr);
    launch_gemm(stream, t0, pp(12), pp(2), tB, R_ * R_, CZ_, 128, 2, pair_mask, k, nullptr);
    trimul_kernel<<<dim3(R_ / 32, R_ / 32, 128), dim3(128), 0, stream>>>(
        tA, tB, t1, R_, 128, dir == 0 ? 1 : 0);
    launch_ln(stream, t1, pp(9), pp(8), v, R_ * R_, 128);
    launch_gemm(stream, t0, pp(14), pp(4), g, R_ * R_, CZ_, CZ_, 2, nullptr, nullptr, nullptr);
    launch_gemm(stream, v, pp(15), pp(5), z_cur, R_ * R_, 128, CZ_, 0, nullptr, g, z_cur);
  }

  // ---------------- Triangle attention (starting node) ----------------
  launch_ln(stream, z_cur, in(TAS_LNG), in(TAS_LNB), t0, R_ * R_, CZ_);
  biasproj_kernel<<<dim3((4 * R_ * R_) / 256), dim3(256), 0, stream>>>(
      t0, in(TAS_WB), pairb, R_, CZ_, 4);
  maskbias_kernel<<<dim3((R_ * R_) / 256), dim3(256), 0, stream>>>(pair_mask, maskb, R_, R_, 0);
  launch_gemm(stream, t0, in(TAS_WQ), nullptr, q, R_ * R_, CZ_, 128, 0, nullptr, nullptr, nullptr, 1);
  launch_gemm(stream, t0, in(TAS_WK), nullptr, k, R_ * R_, CZ_, 128, 0, nullptr, nullptr, nullptr, 1);
  launch_gemm(stream, t0, in(TAS_WV), nullptr, v, R_ * R_, CZ_, 128, 0, nullptr, nullptr, nullptr, 1);
  launch_gemm(stream, t0, in(TAS_WG), in(TAS_BG), g, R_ * R_, CZ_, 128, 0, nullptr, nullptr, nullptr);
  attn_kernel<256><<<dim3(R_, 4), dim3(256), attn_lds(256), stream>>>(
      qb, kb, vb, g, maskb, pairb, tA, 4, SCL);
  launch_gemm(stream, tA, in(TAS_WO), in(TAS_BO), z_cur, R_ * R_, 128, CZ_, 0, nullptr, nullptr, z_cur);

  // ---------------- Triangle attention (ending node) ----------------
  transpose_kernel<<<dim3((unsigned)(ME / 256)), dim3(256), 0, stream>>>(z_cur, t1, R_, R_, CZ_);
  launch_ln(stream, t1, in(TAE_LNG), in(TAE_LNB), t0, R_ * R_, CZ_);
  biasproj_kernel<<<dim3((4 * R_ * R_) / 256), dim3(256), 0, stream>>>(
      t0, in(TAE_WB), pairb, R_, CZ_, 4);
  maskbias_kernel<<<dim3((R_ * R_) / 256), dim3(256), 0, stream>>>(pair_mask, maskb, R_, R_, 1);
  launch_gemm(stream, t0, in(TAE_WQ), nullptr, q, R_ * R_, CZ_, 128, 0, nullptr, nullptr, nullptr, 1);
  launch_gemm(stream, t0, in(TAE_WK), nullptr, k, R_ * R_, CZ_, 128, 0, nullptr, nullptr, nullptr, 1);
  launch_gemm(stream, t0, in(TAE_WV), nullptr, v, R_ * R_, CZ_, 128, 0, nullptr, nullptr, nullptr, 1);
  launch_gemm(stream, t0, in(TAE_WG), in(TAE_BG), g, R_ * R_, CZ_, 128, 0, nullptr, nullptr, nullptr);
  attn_kernel<256><<<dim3(R_, 4), dim3(256), attn_lds(256), stream>>>(
      qb, kb, vb, g, maskb, pairb, tA, 4, SCL);
  launch_gemm(stream, tA, in(TAE_WO), in(TAE_BO), tB, R_ * R_, 128, CZ_, 0, nullptr, nullptr, nullptr);
  addtrans_kernel<<<dim3((unsigned)(ME / 256)), dim3(256), 0, stream>>>(z_cur, tB, R_, R_, CZ_);

  // ---------------- Pair transition ----------------
  launch_ln(stream, z_cur, in(PT_LNG), in(PT_LNB), t0, R_ * R_, CZ_);
  launch_gemm(stream, t0, in(PT_W1), in(PT_B1), hidden, R_ * R_, CZ_, 512, 1, nullptr, nullptr, nullptr);
  launch_gemm(stream, hidden, in(PT_W2), in(PT_B2), z_cur, R_ * R_, 512, CZ_, 0, nullptr, nullptr, z_cur);

  // ---------------- Outputs: (m, z) concatenated ----------------
  hipMemcpyAsync(d_out, m_cur, ME * sizeof(float), hipMemcpyDeviceToDevice, stream);
  hipMemcpyAsync((float*)d_out + ME, z_cur, ME * sizeof(float), hipMemcpyDeviceToDevice, stream);
}
